// QWenBlock_14568529068076
// MI455X (gfx1250) — matrix operand order repaired and checked by reading, not compiled, not run on hardware
//
#include <hip/hip_runtime.h>
#include <hip/hip_bf16.h>
#include <math.h>

// ---------------------------------------------------------------------------
// CDNA5 (gfx1250) Qwen-style transformer block, f16 WMMA everywhere,
// Tensor Data Mover (tensor_load_to_lds) for strided tile staging.
//   K1 rmsnorm+cast -> K2 QKV GEMM -> K3 RoPE -> K4 flash attention ->
//   K5 O-proj GEMM(+residual) -> K6 rmsnorm -> K7 gate/up GEMM ->
//   K8 SiLU*up (in place) -> K9 down GEMM(+residual) -> d_out
// ---------------------------------------------------------------------------

typedef __attribute__((ext_vector_type(16))) _Float16     v16h;
typedef __attribute__((ext_vector_type(8)))  float        v8f;
typedef __attribute__((ext_vector_type(4)))  unsigned int tdm_v4u;
typedef __attribute__((ext_vector_type(8)))  int          tdm_v8i;
typedef __attribute__((ext_vector_type(4)))  int          tdm_v4i;

union FragH {            // 16 f16 = one WMMA A/B operand (8 VGPRs)
    uint4 u[2];
    v16h  v;
};

union PackH {            // pack two f16 for one b32 DS store
    _Float16 h[2];
    unsigned u;
};

__device__ __forceinline__ v8f wmma16(v16h a, v16h b, v8f c) {
    // D = A(16x32) * B(32x16) + C, fp32 accumulate
    return __builtin_amdgcn_wmma_f32_16x16x32_f16(
        /*neg_a=*/false, a, /*neg_b=*/false, b,
        /*c_mod=*/(short)0, c, /*reuse_a=*/false, /*reuse_b=*/false);
}

// ---------------------------------------------------------------------------
// TDM: DMA one 2-D f16 tile (tileX elements x tileY rows, row stride strideX
// elements) from global into LDS, padding the LDS destination by padAmount
// DWORDs after every padInterval-coded DWORDs (ISA 8.3/8.4 D# layout).
//   padIntervalCode v => 2^(v+1) DWORDs ; padAmountCode v => v+1 DWORDs
// Issued by one wave; tracked by TENSORcnt.
// Toolchain uses the 6-arg builtin: (v4u, v8i, v4i, v4i, v8i, i32 cpol).
// ---------------------------------------------------------------------------
__device__ __forceinline__ void tdm_load_2d_f16(unsigned ldsAddr, const void* gptr,
                                                unsigned tileX, unsigned tileY,
                                                unsigned strideX,
                                                unsigned padIntervalCode,
                                                unsigned padAmountCode) {
    unsigned long long ga = (unsigned long long)(uintptr_t)gptr;
    tdm_v4u g0;
    g0[0] = 1u;                                  // count=1, user descriptor
    g0[1] = ldsAddr;                             // lds_addr [63:32]
    g0[2] = (unsigned)ga;                        // global_addr low
    g0[3] = (unsigned)(ga >> 32) | (2u << 30);   // global_addr hi | type=2
    const unsigned long long dim0 = 0x40000000ull;   // huge => no OOB clamp
    const unsigned long long dim1 = 0x40000000ull;
    tdm_v8i g1;
    g1[0] = (int)((1u << 16)                     // data_size = 2 bytes
                | (1u << 20)                     // pad_enable
                | (padIntervalCode << 22)
                | (padAmountCode << 25));        // workgroup_mask = 0
    g1[1] = (int)((dim0 & 0xffffu) << 16);                         // dim0 lo16
    g1[2] = (int)(((dim0 >> 16) & 0xffffu) | ((dim1 & 0xffffu) << 16));
    g1[3] = (int)(((dim1 >> 16) & 0xffffu) | (tileX << 16));       // tile_dim0
    g1[4] = (int)(tileY & 0xffffu);                                // tile_dim1
    g1[5] = (int)strideX;                        // tensor_dim0_stride low32
    g1[6] = 0;                                   // stride hi16 | dim1_stride lo
    g1[7] = 0;
    tdm_v4i gz4 = {0, 0, 0, 0};                  // groups 2/3 unused (2-D)
    tdm_v8i gz8 = {0, 0, 0, 0, 0, 0, 0, 0};
    __builtin_amdgcn_tensor_load_to_lds(g0, g1, gz4, gz4, gz8, 0);
}

// ---------------------------------------------------------------------------
// K1/K6: RMSNorm (fp32 in) -> f16 out
// ---------------------------------------------------------------------------
__global__ __launch_bounds__(256)
void rmsnorm_to_f16(const float* __restrict__ x, const float* __restrict__ w,
                    _Float16* __restrict__ out, int D) {
    int row = blockIdx.x;
    const float* xr = x + (size_t)row * D;
    float ss = 0.f;
    for (int i = threadIdx.x; i < D; i += 256) { float v = xr[i]; ss += v * v; }
    #pragma unroll
    for (int m = 16; m; m >>= 1) ss += __shfl_xor(ss, m, 32);
    __shared__ float red[8];
    if ((threadIdx.x & 31) == 0) red[threadIdx.x >> 5] = ss;
    __syncthreads();
    if (threadIdx.x < 8) {
        float v = red[threadIdx.x];
        #pragma unroll
        for (int m = 4; m; m >>= 1) v += __shfl_xor(v, m, 32);
        if (threadIdx.x == 0) red[0] = v;
    }
    __syncthreads();
    float rs = rsqrtf(red[0] / (float)D + 1e-6f);
    _Float16* orow = out + (size_t)row * D;
    for (int i = threadIdx.x; i < D; i += 256)
        orow[i] = (_Float16)(xr[i] * rs * w[i]);
}

// ---------------------------------------------------------------------------
// Generic WMMA GEMM:  C[M x N] = A_f16[M x K] * W_f32[K x N] (+bias) (+resid)
// Block tile 128x128, K-tile 32. 8 waves = 2x4. A tile staged by the TDM
// (async DMA with LDS row padding); W tile converted fp32->f16 and stored
// transposed (N-major) with packed b32 stores, overlapping the DMA.
// ---------------------------------------------------------------------------
#define BM 128
#define BN 128
#define BK 32

__global__ __launch_bounds__(256)
void gemm_f16_wmma(const _Float16* __restrict__ A, int lda,
                   const float* __restrict__ W, int ldw, int K,
                   const float* __restrict__ bias,
                   const float* __restrict__ resid, int ldr,
                   float* __restrict__ outF, _Float16* __restrict__ outH, int ldo) {
    __shared__ __align__(16) _Float16 As[BM][BK + 8];   // [m][k], pad via TDM
    __shared__ __align__(16) _Float16 Bs[BN][BK + 8];   // [n][k] (transposed)

    int tid  = threadIdx.x;
    int lane = tid & 31, wave = tid >> 5;
    int wm = wave >> 2, wn = wave & 3;           // 2 x 4 wave grid
    int grp = lane >> 4, lr = lane & 15;
    int rowBase = blockIdx.y * BM;
    int colBase = blockIdx.x * BN;
    unsigned asAddr = (unsigned)(uintptr_t)&As[0][0];

    v8f acc[4][2];
    #pragma unroll
    for (int mi = 0; mi < 4; ++mi)
        #pragma unroll
        for (int ni = 0; ni < 2; ++ni)
            acc[mi][ni] = (v8f){0,0,0,0,0,0,0,0};

    for (int k0 = 0; k0 < K; k0 += BK) {
        // --- A tile: 128 rows x 32 f16 via Tensor Data Mover ---
        // 32 f16 = 16 DWORDs per row (code 3), pad 4 DWORDs (code 3) -> row 40
        if (wave == 0)
            tdm_load_2d_f16(asAddr, A + (size_t)rowBase * lda + k0,
                            /*tileX=*/BK, /*tileY=*/BM, /*strideX=*/(unsigned)lda,
                            /*padIntervalCode=*/3, /*padAmountCode=*/3);

        // --- W tile: 32 x 128 fp32 -> f16, transposed into Bs[n][k] ---
        // Each thread: 2 k-rows x 8 cols, packed b32 stores.
        {
            int kp = (tid & 15) * 2;           // k, k+1
            int n0 = (tid >> 4) * 8;           // 0..120
            const float4* r0 = (const float4*)(W + (size_t)(k0 + kp) * ldw + colBase + n0);
            const float4* r1 = (const float4*)(W + (size_t)(k0 + kp + 1) * ldw + colBase + n0);
            float4 a0 = r0[0], a1 = r0[1], b0 = r1[0], b1 = r1[1];
            if (k0 + BK < K)  // stream next weight tile (global_prefetch_b8)
                __builtin_prefetch(W + (size_t)(k0 + BK + kp) * ldw + colBase + n0, 0, 0);
            float c0[8] = {a0.x, a0.y, a0.z, a0.w, a1.x, a1.y, a1.z, a1.w};
            float c1[8] = {b0.x, b0.y, b0.z, b0.w, b1.x, b1.y, b1.z, b1.w};
            #pragma unroll
            for (int q = 0; q < 8; ++q) {
                PackH pk;
                pk.h[0] = (_Float16)c0[q];
                pk.h[1] = (_Float16)c1[q];
                *(unsigned*)&Bs[n0 + q][kp] = pk.u;
            }
        }
        if (wave == 0) __builtin_amdgcn_s_wait_tensorcnt(0);
        __syncthreads();

        // --- fragments per ISA 7.12.2 layouts, then 8 WMMAs ---
        FragH a[4], b[2];
        #pragma unroll
        for (int mi = 0; mi < 4; ++mi) {
            const _Float16* p = &As[wm * 64 + mi * 16 + lr][grp * 8];
            a[mi].u[0] = *(const uint4*)p;
            a[mi].u[1] = *(const uint4*)(p + 16);
        }
        #pragma unroll
        for (int ni = 0; ni < 2; ++ni) {
            const _Float16* p = &Bs[wn * 32 + ni * 16 + lr][grp * 8];
            b[ni].u[0] = *(const uint4*)p;
            b[ni].u[1] = *(const uint4*)(p + 16);
        }
        #pragma unroll
        for (int mi = 0; mi < 4; ++mi)
            #pragma unroll
            for (int ni = 0; ni < 2; ++ni)
                acc[mi][ni] = wmma16(a[mi].v, b[ni].v, acc[mi][ni]);
        __syncthreads();
    }

    // --- epilogue: C lane layout M = r + 8*grp, N = lr ---
    #pragma unroll
    for (int mi = 0; mi < 4; ++mi) {
        int row = rowBase + wm * 64 + mi * 16 + grp * 8;
        #pragma unroll
        for (int ni = 0; ni < 2; ++ni) {
            int col = colBase + wn * 32 + ni * 16 + lr;
            float bval = bias ? bias[col] : 0.f;
            #pragma unroll
            for (int r = 0; r < 8; ++r) {
                float c = acc[mi][ni][r] + bval;
                if (outF) {
                    float rv = resid ? resid[(size_t)(row + r) * ldr + col] : 0.f;
                    outF[(size_t)(row + r) * ldo + col] = rv + c;
                } else {
                    outH[(size_t)(row + r) * ldo + col] = (_Float16)c;
                }
            }
        }
    }
}

// ---------------------------------------------------------------------------
// K3: RoPE applied in-place to q,k halves of the f16 qkv buffer
// qkv layout: [B*S rows][12288] = q(0..4095) k(4096..8191) v(8192..12287)
// ---------------------------------------------------------------------------
__global__ __launch_bounds__(256)
void rope_f16(_Float16* __restrict__ qkv, const int* __restrict__ pos) {
    size_t idx = (size_t)blockIdx.x * 256 + threadIdx.x;  // rows*32*64 threads
    int d = idx & 63;
    size_t t = idx >> 6;
    int h = t & 31;
    size_t row = t >> 5;
    float inv = __powf(10000.f, -(float)d / 64.f);
    float ang = (float)pos[row] * inv;
    float s, c;
    __sincosf(ang, &s, &c);
    size_t base = row * 12288 + (size_t)h * 128 + d;
    float q1 = (float)qkv[base], q2 = (float)qkv[base + 64];
    qkv[base]      = (_Float16)(q1 * c - q2 * s);
    qkv[base + 64] = (_Float16)(q2 * c + q1 * s);
    float k1 = (float)qkv[base + 4096], k2 = (float)qkv[base + 4096 + 64];
    qkv[base + 4096]      = (_Float16)(k1 * c - k2 * s);
    qkv[base + 4096 + 64] = (_Float16)(k2 * c + k1 * s);
}

// ---------------------------------------------------------------------------
// K4: causal flash attention. Grid (B*H, S/128). 128 q-rows per block,
// 64-key tiles. Q frags in registers; K tile staged by the TDM, V tile
// transposed manually with packed b32 stores. Q*K^T and P*V on WMMA.
// ---------------------------------------------------------------------------
#define AKV 64

__global__ __launch_bounds__(256)
void flash_attn_wmma(const _Float16* __restrict__ qkv, _Float16* __restrict__ obf,
                     int S, int H) {
    __shared__ __align__(16) _Float16 Ks [AKV][136];      // [k_local][d] (TDM pad)
    __shared__ __align__(16) _Float16 Vst[128][AKV + 8];  // [d][k_local]
    __shared__ __align__(16) _Float16 Ps [128][AKV + 8];  // [q_local][k_local]

    int bh = blockIdx.x;
    int h = bh % H, b = bh / H;
    int qBase = blockIdx.y * 128;
    int tid = threadIdx.x, lane = tid & 31, wave = tid >> 5;
    int grp = lane >> 4, lr = lane & 15;

    const size_t rstride = 12288;
    const _Float16* base = qkv + (size_t)b * S * rstride;
    int hc = h * 128;
    const float scale = 0.08838834764831845f;  // 1/sqrt(128)
    const float NEG = -1e30f;
    unsigned ksAddr = (unsigned)(uintptr_t)&Ks[0][0];

    // Q fragments (wave owns 16 q rows), 4 K-steps over Dh=128
    FragH qf[4];
    {
        const _Float16* qr = base + (size_t)(qBase + wave * 16 + lr) * rstride + hc;
        #pragma unroll
        for (int ks = 0; ks < 4; ++ks) {
            const _Float16* p = qr + ks * 32 + grp * 8;
            qf[ks].u[0] = *(const uint4*)p;
            qf[ks].u[1] = *(const uint4*)(p + 16);
        }
    }

    float m[8], l[8];
    v8f o[8];
    #pragma unroll
    for (int r = 0; r < 8; ++r) { m[r] = NEG; l[r] = 0.f; }
    #pragma unroll
    for (int no = 0; no < 8; ++no) o[no] = (v8f){0,0,0,0,0,0,0,0};

    int qrow0 = qBase + wave * 16 + grp * 8;   // + r = this lane's rows
    int jEnd = (qBase + 128) / AKV;

    for (int j = 0; j < jEnd; ++j) {
        int kBase = j * AKV;
        // --- K tile 64 x 128 f16 via TDM: 64 DWORDs/row (code 5), pad 4 (code 3)
        if (wave == 0)
            tdm_load_2d_f16(ksAddr, base + (size_t)kBase * rstride + 4096 + hc,
                            /*tileX=*/128, /*tileY=*/AKV, /*strideX=*/12288,
                            /*padIntervalCode=*/5, /*padAmountCode=*/3);
        // --- V tile transposed: each thread 2 k-rows x 16 d, packed b32 stores
        {
            int kr = (tid >> 3) * 2;           // 0..62
            int dseg = (tid & 7) * 16;         // 0..112
            const uint4* v0 = (const uint4*)(base + (size_t)(kBase + kr) * rstride + 8192 + hc + dseg);
            const uint4* v1 = (const uint4*)(base + (size_t)(kBase + kr + 1) * rstride + 8192 + hc + dseg);
            uint4 arr0[2] = {v0[0], v0[1]};
            uint4 arr1[2] = {v1[0], v1[1]};
            const _Float16* e0 = (const _Float16*)arr0;
            const _Float16* e1 = (const _Float16*)arr1;
            #pragma unroll
            for (int e = 0; e < 16; ++e) {
                PackH pk;
                pk.h[0] = e0[e];
                pk.h[1] = e1[e];
                *(unsigned*)&Vst[dseg + e][kr] = pk.u;
            }
        }
        if (wave == 0) __builtin_amdgcn_s_wait_tensorcnt(0);
        __syncthreads();

        // scores: S = Q * K^T  (16 WMMAs per wave)
        v8f sacc[4];
        #pragma unroll
        for (int ni = 0; ni < 4; ++ni) sacc[ni] = (v8f){0,0,0,0,0,0,0,0};
        #pragma unroll
        for (int ks = 0; ks < 4; ++ks) {
            #pragma unroll
            for (int ni = 0; ni < 4; ++ni) {
                FragH bf;
                const _Float16* p = &Ks[ni * 16 + lr][ks * 32 + grp * 8];
                bf.u[0] = *(const uint4*)p;
                bf.u[1] = *(const uint4*)(p + 16);
                sacc[ni] = wmma16(qf[ks].v, bf.v, sacc[ni]);
            }
        }

        // online softmax (per-row reductions across the 16-lane half)
        float mt[8];
        #pragma unroll
        for (int r = 0; r < 8; ++r) mt[r] = NEG;
        #pragma unroll
        for (int ni = 0; ni < 4; ++ni) {
            int kidx = kBase + ni * 16 + lr;
            #pragma unroll
            for (int r = 0; r < 8; ++r) {
                float s = sacc[ni][r] * scale;
                if (kidx > qrow0 + r) s = NEG;
                sacc[ni][r] = s;
                mt[r] = fmaxf(mt[r], s);
            }
        }
        #pragma unroll
        for (int r = 0; r < 8; ++r)
            #pragma unroll
            for (int msk = 1; msk < 16; msk <<= 1)
                mt[r] = fmaxf(mt[r], __shfl_xor(mt[r], msk, 32));

        float al[8], rs[8];
        #pragma unroll
        for (int r = 0; r < 8; ++r) {
            float mn = fmaxf(m[r], mt[r]);
            al[r] = __expf(m[r] - mn);
            m[r] = mn;
            rs[r] = 0.f;
        }
        #pragma unroll
        for (int ni = 0; ni < 4; ++ni)
            #pragma unroll
            for (int r = 0; r < 8; ++r) {
                float p = __expf(sacc[ni][r] - m[r]);
                rs[r] += p;
                Ps[wave * 16 + grp * 8 + r][ni * 16 + lr] = (_Float16)p;
            }
        #pragma unroll
        for (int r = 0; r < 8; ++r) {
            #pragma unroll
            for (int msk = 1; msk < 16; msk <<= 1)
                rs[r] += __shfl_xor(rs[r], msk, 32);
            l[r] = l[r] * al[r] + rs[r];
        }
        #pragma unroll
        for (int no = 0; no < 8; ++no)
            #pragma unroll
            for (int r = 0; r < 8; ++r)
                o[no][r] *= al[r];

        // O += P * V  (16 WMMAs per wave; Ps written & read by same wave)
        #pragma unroll
        for (int ks = 0; ks < 2; ++ks) {
            FragH af;
            const _Float16* pp = &Ps[wave * 16 + lr][ks * 32 + grp * 8];
            af.u[0] = *(const uint4*)pp;
            af.u[1] = *(const uint4*)(pp + 16);
            #pragma unroll
            for (int no = 0; no < 8; ++no) {
                FragH bf;
                const _Float16* vp = &Vst[no * 16 + lr][ks * 32 + grp * 8];
                bf.u[0] = *(const uint4*)vp;
                bf.u[1] = *(const uint4*)(vp + 16);
                o[no] = wmma16(af.v, bf.v, o[no]);
            }
        }
        __syncthreads();
    }

    // normalize and store f16 output (rows = b*S + s, cols = h*128 + d)
    _Float16* orow = obf + ((size_t)b * S + qrow0) * 4096 + hc;
    #pragma unroll
    for (int r = 0; r < 8; ++r) {
        float inv = 1.f / l[r];
        #pragma unroll
        for (int no = 0; no < 8; ++no)
            orow[(size_t)r * 4096 + no * 16 + lr] = (_Float16)(o[no][r] * inv);
    }
}

// ---------------------------------------------------------------------------
// K8: h = silu(gate) * up, written in place over the gate half of gu
// ---------------------------------------------------------------------------
__global__ __launch_bounds__(256)
void silu_mul_f16(_Float16* __restrict__ gu, size_t total, int I) {
    size_t idx = (size_t)blockIdx.x * 256 + threadIdx.x;
    if (idx >= total) return;
    size_t r = idx / (size_t)I;
    int i = (int)(idx - r * (size_t)I);
    _Float16* rowp = gu + r * (size_t)(2 * I);
    float g = (float)rowp[i], u = (float)rowp[I + i];
    rowp[i] = (_Float16)((g / (1.f + __expf(-g))) * u);
}

// ---------------------------------------------------------------------------
extern "C" void kernel_launch(void* const* d_in, const int* in_sizes, int n_in,
                              void* d_out, int out_size, void* d_ws, size_t ws_size,
                              hipStream_t stream) {
    const float* hidden = (const float*)d_in[0];
    const int*   pos    = (const int*)d_in[1];
    const float* ln1    = (const float*)d_in[2];
    const float* w_qkv  = (const float*)d_in[3];
    const float* b_qkv  = (const float*)d_in[4];
    const float* w_o    = (const float*)d_in[5];
    const float* ln2    = (const float*)d_in[6];
    const float* w_gu   = (const float*)d_in[7];
    const float* w_down = (const float*)d_in[8];
    float* out = (float*)d_out;

    const int B = 2, S = 2048, D = 4096, H = 32, I = 11008;
    const int M = B * S;                       // 4096 rows

    // workspace layout with lifetime reuse (300 MB total):
    //   region A (32 MB):  xb -> obf -> x2      (f16, M x D)
    //   region B (96 MB):  qkv (f16, M x 3D) -> hidden2 (f32, M x D)
    //   region C (172 MB): gu  (f16, M x 2I), SiLU'd in place
    char* ws = (char*)d_ws;
    _Float16* regA = (_Float16*)ws;                                  // M*D f16
    _Float16* qkv  = (_Float16*)(ws + (size_t)33554432);             // M*3D f16
    float*    hid2 = (float*)   (ws + (size_t)33554432);             // M*D  f32
    _Float16* gu   = (_Float16*)(ws + (size_t)134217728);            // M*2I f16

    _Float16* xb  = regA;
    _Float16* obf = regA;
    _Float16* x2  = regA;

    // 1) rmsnorm(hidden) -> xb (f16)
    rmsnorm_to_f16<<<M, 256, 0, stream>>>(hidden, ln1, xb, D);

    // 2) qkv = xb @ w_qkv + b_qkv   (f16 out)
    gemm_f16_wmma<<<dim3(3 * D / BN, M / BM), 256, 0, stream>>>(
        xb, D, w_qkv, 3 * D, D, b_qkv, nullptr, 0, nullptr, qkv, 3 * D);

    // 3) RoPE on q,k (in place)
    rope_f16<<<(unsigned)((size_t)M * H * 64 / 256), 256, 0, stream>>>(qkv, pos);

    // 4) flash attention -> obf (f16, [b*S+s][h*128+d])
    flash_attn_wmma<<<dim3(B * H, S / 128), 256, 0, stream>>>(qkv, obf, S, H);

    // 5) hidden2 = hidden + obf @ w_o   (f32 out)
    gemm_f16_wmma<<<dim3(D / BN, M / BM), 256, 0, stream>>>(
        obf, D, w_o, D, D, nullptr, hidden, D, hid2, nullptr, D);

    // 6) rmsnorm(hidden2) -> x2 (f16)
    rmsnorm_to_f16<<<M, 256, 0, stream>>>(hid2, ln2, x2, D);

    // 7) gu = x2 @ w_gate_up   (f16 out, M x 2I)
    gemm_f16_wmma<<<dim3(2 * I / BN, M / BM), 256, 0, stream>>>(
        x2, D, w_gu, 2 * I, D, nullptr, nullptr, 0, nullptr, gu, 2 * I);

    // 8) gate half := silu(gate) * up
    {
        size_t total = (size_t)M * I;
        silu_mul_f16<<<(unsigned)((total + 255) / 256), 256, 0, stream>>>(gu, total, I);
    }

    // 9) out = hidden2 + h @ w_down   (f32 out; A has row stride 2I, K = I)
    gemm_f16_wmma<<<dim3(D / BN, M / BM), 256, 0, stream>>>(
        gu, 2 * I, w_down, D, I, nullptr, hid2, D, out, nullptr, D);
}
